// PPOAgent_27101243638254
// MI455X (gfx1250) — matrix operand order repaired and checked by reading, not compiled, not run on hardware
//
#include <hip/hip_runtime.h>
#include <stdint.h>

#define CIN 4
#define LEN 256
#define C1 16
#define C2 32
#define C3 64
#define HIDDEN 64

#define N_TILES 17                      // 1 (conv1) + 4 (conv2) + 12 (conv3)

typedef __attribute__((ext_vector_type(16))) __bf16        v16bf;
typedef __attribute__((ext_vector_type(8)))  float         v8f;
typedef __attribute__((ext_vector_type(4)))  float         v4f;
typedef __attribute__((ext_vector_type(4)))  unsigned int  v4u;

// POD union: build/load WMMA operands via b128 ops.
union BfTile {
    unsigned short h[16];
    __bf16         b[16];
    unsigned int   u[8];
    v4u            q[2];
    v16bf          v;
};

// One-instruction packed bf16 convert (RNE). The compiler emits this mnemonic
// itself for gfx1250, but only in unpaired form — force the paired form.
static __device__ __forceinline__ unsigned int pack_bf16(float lo, float hi) {
    unsigned int r;
    asm("v_cvt_pk_bf16_f32 %0, %1, %2" : "=v"(r) : "v"(lo), "v"(hi));
    return r;
}

// One-instruction ReLU (avoids clang's canonicalize + max pair).
static __device__ __forceinline__ float relu(float f) {
    float r;
    asm("v_max_num_f32 %0, 0, %1" : "=v"(r) : "v"(f));
    return r;
}

// A-tile table lives in a device global; prep kernel rewrites it every launch.
__device__ __align__(16) unsigned int g_atiles[N_TILES * 32 * 8];

// Build WMMA A-tile `t` for one lane (contiguous 16-bit A 16x32 layout:
// element e -> K = e + 16*khalf ; M = lane&15).
// Tile map: 0 = conv1 ; 1..4 = conv2[mt][ks] ; 5..16 = conv3[mt][tap].
__global__ void build_atiles_kernel(const float* __restrict__ w1,
                                    const float* __restrict__ w2,
                                    const float* __restrict__ w3) {
    const int idx = threadIdx.x;            // one block of N_TILES*32 threads
    if (idx >= N_TILES * 32) return;
    const int t = idx >> 5, lane = idx & 31;
    const int mrow = lane & 15, khalf = lane >> 4;

    BfTile tile;
#pragma unroll
    for (int e = 0; e < 16; ++e) {
        const int kk = e + khalf * 16;   // K index 0..31
        float wv = 0.f;
        if (t == 0) {                       // conv1: K = tap*4 + cin (valid < 12)
            if (kk < 12) wv = w1[mrow * 12 + (kk & 3) * 3 + (kk >> 2)];
        } else if (t < 5) {                 // conv2
            const int mt = (t - 1) >> 1, ks = (t - 1) & 1;
            const int co = mt * 16 + mrow;
            if (ks == 0)        wv = w2[co * 48 + (kk & 15) * 3 + (kk >> 4)];
            else if (kk < 16)   wv = w2[co * 48 + kk * 3 + 2];
        } else {                            // conv3: K = cin, one tap per tile
            const int q = t - 5, mt = q / 3, tap = q % 3;
            wv = w3[(mt * 16 + mrow) * 96 + kk * 3 + tap];
        }
        tile.b[e] = (__bf16)wv;
    }
    v4u* dst = (v4u*)(g_atiles + idx * 8);
    dst[0] = tile.q[0];
    dst[1] = tile.q[1];
}

__global__ __launch_bounds__(256) void fused_cnn1d_kernel(
    const float* __restrict__ x,
    const float* __restrict__ b1,
    const float* __restrict__ b2,
    const float* __restrict__ b3,
    const float* __restrict__ fc1w, const float* __restrict__ fc1b,
    const float* __restrict__ fc2w, const float* __restrict__ fc2b,
    float* __restrict__ out)
{
    // LDS staging: all activations on-chip (~37.6 KB).
    __shared__ float xs[CIN][LEN + 2];                          // fp32 input + halo
    __shared__ __align__(16) unsigned int xbu[LEN][8];          // im2col bf16 [pos][K=12 pad 16]
    __shared__ __align__(16) unsigned int h1u[LEN + 2][C1 / 2]; // bf16 [pos][cin]
    __shared__ __align__(16) unsigned int h2u[LEN + 2][C2 / 2]; // bf16 [pos][cin]
    __shared__ float chanSum[C3];
    __shared__ float hid[HIDDEN];

    const int tid   = threadIdx.x;
    const int lane  = tid & 31;
    const int wave  = tid >> 5;
    const int s     = blockIdx.x;
    const int mrow  = lane & 15;
    const int khalf = lane >> 4;
    const int posBase = wave * 32;          // each wave owns 2 N-tiles

    // ---- zero halos / accumulators -------------------------------------
    if (tid < CIN)    { xs[tid][0] = 0.f; xs[tid][LEN + 1] = 0.f; }
    if (tid < C1 / 2) { h1u[0][tid] = 0u; h1u[LEN + 1][tid] = 0u; }
    if (tid < C2 / 2) { h2u[0][tid] = 0u; h2u[LEN + 1][tid] = 0u; }
    if (tid < C3)     { chanSum[tid] = 0.f; }

    // ---- stage input sample into LDS -----------------------------------
    {
        const float* xp = x + (size_t)s * (CIN * LEN);
        __builtin_prefetch(xp + CIN * LEN, 0, 1);   // global_prefetch_b8
#pragma unroll
        for (int c = 0; c < CIN; ++c)
            xs[c][tid + 1] = xp[c * LEN + tid];
    }

    // ---- fetch precomputed A-tiles (2x global_load_b128 each) ----------
    BfTile a1, a2[2][2], a3[4][3];
    {
        const v4u* at = (const v4u*)g_atiles;
        const int base = lane * 2;
        a1.q[0] = at[base];      a1.q[1] = at[base + 1];
#pragma unroll
        for (int mt = 0; mt < 2; ++mt)
#pragma unroll
            for (int ks = 0; ks < 2; ++ks) {
                const int t = 1 + mt * 2 + ks;
                a2[mt][ks].q[0] = at[t * 64 + base];
                a2[mt][ks].q[1] = at[t * 64 + base + 1];
            }
#pragma unroll
        for (int mt = 0; mt < 4; ++mt)
#pragma unroll
            for (int tp = 0; tp < 3; ++tp) {
                const int t = 5 + mt * 3 + tp;
                a3[mt][tp].q[0] = at[t * 64 + base];
                a3[mt][tp].q[1] = at[t * 64 + base + 1];
            }
    }

    // ---- per-lane bias vectors (become WMMA C inputs) ------------------
    v8f bc1, bc2[2], bc3[4];
    {
        const v4f* p = (const v4f*)(b1 + khalf * 8);
#pragma unroll
        for (int j = 0; j < 4; ++j) { bc1[j] = p[0][j]; bc1[4 + j] = p[1][j]; }
#pragma unroll
        for (int mt = 0; mt < 2; ++mt) {
            const v4f* q = (const v4f*)(b2 + mt * 16 + khalf * 8);
#pragma unroll
            for (int j = 0; j < 4; ++j) { bc2[mt][j] = q[0][j]; bc2[mt][4 + j] = q[1][j]; }
        }
#pragma unroll
        for (int mt = 0; mt < 4; ++mt) {
            const v4f* q = (const v4f*)(b3 + mt * 16 + khalf * 8);
#pragma unroll
            for (int j = 0; j < 4; ++j) { bc3[mt][j] = q[0][j]; bc3[mt][4 + j] = q[1][j]; }
        }
    }

    __syncthreads();

    // ---- build conv1 im2col in bf16: xb[pos][k], k = tap*4+cin (12, pad 16)
    {
        float v[12];
#pragma unroll
        for (int k = 0; k < 12; ++k)
            v[k] = xs[k & 3][tid + (k >> 2)];   // halo handles boundaries
#pragma unroll
        for (int j = 0; j < 6; ++j)
            xbu[tid][j] = pack_bf16(v[2 * j], v[2 * j + 1]);
        xbu[tid][6] = 0u; xbu[tid][7] = 0u;     // K 12..15 zero
    }
    __syncthreads();

    // ---- conv1 (4->16) : WMMA, K=12 (A upper half zero) ----------------
#pragma unroll
    for (int pt = 0; pt < 2; ++pt) {
        const int p0 = posBase + pt * 16;
        BfTile bx;
        const v4u* src = (const v4u*)&xbu[p0 + mrow][0];
        bx.q[0] = src[0]; bx.q[1] = src[1];
        v8f c = bc1;
        c = __builtin_amdgcn_wmma_f32_16x16x32_bf16(false, a1.v, false, bx.v,
                                                    (short)0, c, false, false);
        v4u p;
#pragma unroll
        for (int j = 0; j < 4; ++j)
            p[j] = pack_bf16(relu(c[2 * j]), relu(c[2 * j + 1]));
        *(v4u*)&h1u[p0 + mrow + 1][khalf * 4] = p;
    }
    __syncthreads();

    // ---- conv2 (16->32) : WMMA, K=48 -> 2x32 ---------------------------
    {
        v8f d2[2][2];
#pragma unroll
        for (int pt = 0; pt < 2; ++pt) {
            const int p0 = posBase + pt * 16;
            BfTile b0, bt1;
            {   // k-step0: lanes0-15 tap0 (row p0+n), lanes16-31 tap1 (row p0+n+1)
                const v4u* src = (const v4u*)&h1u[p0 + mrow + khalf][0];
                b0.q[0] = src[0]; b0.q[1] = src[1];
                // k-step1: lanes0-15 tap2; lanes16-31 don't-care (A half zero)
                const v4u* src1 = (const v4u*)&h1u[p0 + mrow + (khalf ? 0 : 2)][0];
                bt1.q[0] = src1[0]; bt1.q[1] = src1[1];
            }
#pragma unroll
            for (int mt = 0; mt < 2; ++mt) {
                v8f c = bc2[mt];
                c = __builtin_amdgcn_wmma_f32_16x16x32_bf16(false, a2[mt][0].v,
                        false, b0.v, (short)0, c, false, false);
                c = __builtin_amdgcn_wmma_f32_16x16x32_bf16(false, a2[mt][1].v,
                        false, bt1.v, (short)0, c, false, false);
                d2[pt][mt] = c;
            }
        }
#pragma unroll
        for (int pt = 0; pt < 2; ++pt) {
            const int row = posBase + pt * 16 + mrow + 1;
#pragma unroll
            for (int mt = 0; mt < 2; ++mt) {
                v4u p;
#pragma unroll
                for (int j = 0; j < 4; ++j)
                    p[j] = pack_bf16(relu(d2[pt][mt][2 * j]),
                                     relu(d2[pt][mt][2 * j + 1]));
                *(v4u*)&h2u[row][mt * 8 + khalf * 4] = p;
            }
        }
    }
    __syncthreads();

    // ---- conv3 (32->64) : WMMA, K=96 -> 3x32, then mean over L ---------
    {
        v8f d3[2][4];
#pragma unroll
        for (int pt = 0; pt < 2; ++pt) {
            const int p0 = posBase + pt * 16;
            BfTile bt[3];
#pragma unroll
            for (int t = 0; t < 3; ++t) {
                const v4u* src = (const v4u*)&h2u[p0 + mrow + t][khalf * 8];
                bt[t].q[0] = src[0]; bt[t].q[1] = src[1];
            }
#pragma unroll
            for (int mt = 0; mt < 4; ++mt) {
                v8f c = bc3[mt];    // bias per position-tile accumulator
#pragma unroll
                for (int t = 0; t < 3; ++t)
                    c = __builtin_amdgcn_wmma_f32_16x16x32_bf16(false, a3[mt][t].v,
                            false, bt[t].v, (short)0, c, false, false);
                d3[pt][mt] = c;
            }
        }
        // ReLU per position, fold the two position-tiles, one butterfly per (mt,r)
#pragma unroll
        for (int mt = 0; mt < 4; ++mt) {
#pragma unroll
            for (int r = 0; r < 8; ++r) {
                float v = relu(d3[0][mt][r]) + relu(d3[1][mt][r]);
                v += __shfl_xor(v, 1, 32);
                v += __shfl_xor(v, 2, 32);
                v += __shfl_xor(v, 4, 32);
                v += __shfl_xor(v, 8, 32);   // stays within 16-lane half
                if (mrow == 0)
                    atomicAdd(&chanSum[mt * 16 + khalf * 8 + r], v);
            }
        }
    }
    __syncthreads();

    // ---- mean + MLP (64->64 ReLU, 64->1) -------------------------------
    if (tid < HIDDEN) {
        float acc = fc1b[tid];
#pragma unroll 8
        for (int c = 0; c < C3; ++c)
            acc = fmaf(fc1w[tid * C3 + c], chanSum[c] * (1.0f / LEN), acc);
        hid[tid] = relu(acc);
    }
    __syncthreads();
    if (tid == 0) {
        float acc = fc2b[0];
#pragma unroll 8
        for (int h = 0; h < HIDDEN; ++h)
            acc = fmaf(fc2w[h], hid[h], acc);
        out[s] = acc;
    }
}

extern "C" void kernel_launch(void* const* d_in, const int* in_sizes, int n_in,
                              void* d_out, int out_size, void* d_ws, size_t ws_size,
                              hipStream_t stream) {
    const float* x    = (const float*)d_in[0];
    const float* w1   = (const float*)d_in[1];
    const float* b1   = (const float*)d_in[2];
    const float* w2   = (const float*)d_in[3];
    const float* b2   = (const float*)d_in[4];
    const float* w3   = (const float*)d_in[5];
    const float* b3   = (const float*)d_in[6];
    const float* fc1w = (const float*)d_in[7];
    const float* fc1b = (const float*)d_in[8];
    const float* fc2w = (const float*)d_in[9];
    const float* fc2b = (const float*)d_in[10];
    float* out = (float*)d_out;
    (void)d_ws; (void)ws_size; (void)n_in; (void)out_size;

    const int batch = in_sizes[0] / (CIN * LEN);   // 16384
    if (batch <= 0) return;

    // Bake grid-invariant WMMA A-tiles (rewritten every call; deterministic).
    build_atiles_kernel<<<1, N_TILES * 32, 0, stream>>>(w1, w2, w3);

    fused_cnn1d_kernel<<<batch, 256, 0, stream>>>(
        x, b1, b2, b3, fc1w, fc1b, fc2w, fc2b, out);
}
